// MixtralBlock_72851235275309
// MI455X (gfx1250) — matrix operand order repaired and checked by reading, not compiled, not run on hardware
//
#include <hip/hip_runtime.h>
#include <hip/hip_bf16.h>
#include <cstdint>

// ---------------- types ----------------
typedef __attribute__((ext_vector_type(16))) __bf16       v16bf;
typedef __attribute__((ext_vector_type(4)))  __bf16       v4bf;
typedef __attribute__((ext_vector_type(8)))  float        v8f;
typedef __attribute__((ext_vector_type(4)))  unsigned int v4u;
typedef __attribute__((ext_vector_type(4)))  unsigned int u32x4;
typedef __attribute__((ext_vector_type(8)))  int          i32x8;
typedef __attribute__((ext_vector_type(4)))  int          i32x4;

// ---------------- problem dims ----------------
constexpr int kBatch = 2;
constexpr int kT     = 1024;
constexpr int kD     = 1024;
constexpr int kH     = 16;
constexpr int kHD    = 64;
constexpr int kE     = 8;
constexpr int kDFF   = 4096;
constexpr int kTok   = kBatch * kT;   // 2048
constexpr int kRows  = kTok * 2;      // 4096 gathered MoE rows (top-2)
constexpr int kBand  = 512;

#if __has_builtin(__builtin_amdgcn_tensor_load_to_lds)
#define USE_TDM 1
#else
#define USE_TDM 0
#endif

// ---------------- helpers ----------------
__device__ inline v8f vzero() {
  v8f z = {0.f,0.f,0.f,0.f,0.f,0.f,0.f,0.f};
  return z;
}

struct Frag32 { v4u a, b; };

// A-style fragment: row = lane&15, K elems {kb..kb+8, kb+16..kb+24}, kb=(lane>>4)*8
__device__ inline v16bf ld_fragA(const __bf16* base, int ld, int lane) {
  const __bf16* p = base + (lane & 15) * ld + ((lane >> 4) << 3);
  Frag32 f;
  f.a = *(const v4u*)(p);
  f.b = *(const v4u*)(p + 16);
  return __builtin_bit_cast(v16bf, f);
}

// B-style fragment (represents U^T): row n = lane&15, K elems {kb..kb+8, kb+16..kb+24}, kb=(lane>>4)*8
__device__ inline v16bf ld_fragB(const __bf16* base, int ld, int lane) {
  const __bf16* p = base + (lane & 15) * ld + ((lane >> 4) << 3);
  Frag32 f;
  f.a = *(const v4u*)(p);
  f.b = *(const v4u*)(p + 16);
  return __builtin_bit_cast(v16bf, f);
}

__device__ inline v8f wmma_bf16(v16bf a, v16bf b, v8f c) {
  return __builtin_amdgcn_wmma_f32_16x16x32_bf16(false, a, false, b, (short)0, c,
                                                 false, false);
}

// ---------------- fp32 -> bf16 conversion ----------------
__global__ __launch_bounds__(256) void k_cvt(const float* __restrict__ in,
                                             __bf16* __restrict__ out, int n) {
  int i = (blockIdx.x * 256 + threadIdx.x) * 4;
  if (i >= n) return;
  const float4 f = *(const float4*)(in + i);
  v4bf o;
  o[0] = (__bf16)f.x; o[1] = (__bf16)f.y; o[2] = (__bf16)f.z; o[3] = (__bf16)f.w;
  *(v4bf*)(out + i) = o;
}

// ---------------- RMSNorm (+optional RoPE) ----------------
template <bool ROPE>
__global__ __launch_bounds__(256) void k_rms(const float* __restrict__ x,
                                             const float* __restrict__ scale,
                                             __bf16* __restrict__ outN,
                                             __bf16* __restrict__ outR) {
  const int row = blockIdx.x;
  const float* xr = x + (long long)row * kD;
  float s = 0.f;
  for (int i = threadIdx.x; i < kD; i += 256) { float v = xr[i]; s += v * v; }
  __shared__ float red[256];
  red[threadIdx.x] = s;
  __syncthreads();
  for (int o = 128; o > 0; o >>= 1) {
    if (threadIdx.x < o) red[threadIdx.x] += red[threadIdx.x + o];
    __syncthreads();
  }
  const float inv = rsqrtf(red[0] * (1.f / kD) + 1e-6f);
  const int t = row & (kT - 1);
  for (int i = threadIdx.x * 2; i < kD; i += 512) {
    float a = xr[i] * inv * scale[i];
    float b = xr[i + 1] * inv * scale[i + 1];
    outN[(long long)row * kD + i]     = (__bf16)a;
    outN[(long long)row * kD + i + 1] = (__bf16)b;
    if (ROPE) {
      int d = i & (kHD - 1);
      int pr = d >> 1;
      float theta = powf(10000.f, -(float)(2 * pr) * (1.f / kHD));
      float ang = (float)t * theta;
      float sn, cs;
      sincosf(ang, &sn, &cs);
      outR[(long long)row * kD + i]     = (__bf16)(a * cs - b * sn);
      outR[(long long)row * kD + i + 1] = (__bf16)(b * cs + a * sn);
    }
  }
}

// ---------------- TDM weight-tile load (128 rows x 32 cols bf16, padded) ----
__device__ inline void tdm_w_tile(const __bf16* gptr, void* ldsPtr, int Kd, int Nrows) {
#if USE_TDM
  unsigned long long ga = (unsigned long long)(const void*)gptr;
  unsigned ldsOff = (unsigned)(unsigned long long)ldsPtr;
  u32x4 g0;
  g0[0] = 1u;                                   // count=1
  g0[1] = ldsOff;                               // lds_addr
  g0[2] = (unsigned)ga;                         // global_addr[31:0]
  g0[3] = (unsigned)((ga >> 32) & 0x01FFFFFFull) | (2u << 30);  // addr[56:32] | type=2
  i32x8 g1;
  // data_size=2B, pad_enable, pad_interval=16DW(64B), pad_amount=4DW(16B) -> 80B stride
  g1[0] = (int)((1u << 16) | (1u << 20) | (3u << 22) | (3u << 25));
  g1[1] = (int)(((unsigned)Kd & 0xFFFFu) << 16);                         // dim0 lo
  g1[2] = (int)((((unsigned)Kd) >> 16) | (((unsigned)Nrows & 0xFFFFu) << 16));
  g1[3] = (int)((((unsigned)Nrows) >> 16) | (32u << 16));                // tile_dim0=32
  g1[4] = 128;                                                          // tile_dim1=128
  g1[5] = Kd;                                                           // dim0 stride
  g1[6] = 0;
  g1[7] = 0;
  i32x4 gz4; gz4[0] = gz4[1] = gz4[2] = gz4[3] = 0;
  i32x8 gz8;
#pragma unroll
  for (int i = 0; i < 8; ++i) gz8[i] = 0;
  // 6-arg toolchain variant: (g0, g1, g2, g3, g4, cpol)
  __builtin_amdgcn_tensor_load_to_lds(g0, g1, gz4, gz4, gz8, 0);
  __builtin_amdgcn_s_wait_tensorcnt((short)0);
#else
  (void)gptr; (void)ldsPtr; (void)Kd; (void)Nrows;
#endif
}

// ---------------- generic bf16 WMMA GEMM: C = A * W^T (+bias) ----------------
// MODE 0: store fp32 [M,N]
// MODE 1: QKV store   (N=2048: cols<1024 -> q[b,h,t,hd], else k[b,h,t,hd])
// MODE 2: V store transposed -> vt[b,h,hd,t]
// MODE 3: MoE GEMM1, gathered rows via row_map, SiLU, bf16 h[slot, DFF]
// MODE 4: MoE GEMM2, rows = slots, fp32 y[slot, D]
template <int MODE>
__global__ __launch_bounds__(256) void k_gemm(
    const __bf16* __restrict__ A, const __bf16* __restrict__ W,
    const float* __restrict__ bias,
    float* __restrict__ outF, __bf16* __restrict__ outB, __bf16* __restrict__ outB2,
    int M, int N, int Kd,
    const int* __restrict__ row_map,
    const int* __restrict__ ex_base, const int* __restrict__ ex_cnt,
    long long wStride) {
  const int e  = blockIdx.z;
  const int m0 = blockIdx.x * 128;
  const int n0 = blockIdx.y * 128;
  int Me = M, base = 0;
  if (MODE >= 3) {
    Me = ex_cnt[e];
    base = ex_base[e];
    W += (long long)e * wStride;
    bias += (long long)e * N;
  }
  if (m0 >= Me) return;

  __shared__ __bf16 smA[128 * 40];
  __shared__ __bf16 smW[128 * 40];

  const int tid  = threadIdx.x;
  const int lane = tid & 31;
  const int wv   = tid >> 5;
  const int wm   = (wv & 3) * 32;   // 4 waves down M
  const int wn   = (wv >> 2) * 64;  // 2 waves across N
  v8f acc[2][4];
#pragma unroll
  for (int i = 0; i < 2; ++i)
#pragma unroll
    for (int j = 0; j < 4; ++j) acc[i][j] = vzero();

  for (int k0 = 0; k0 < Kd; k0 += 32) {
    __syncthreads();
    // A tile (gathered for MoE), 128x32 bf16
#pragma unroll
    for (int u = tid; u < 512; u += 256) {
      int r = u >> 2, c4 = u & 3;
      int rr = m0 + r;
      int ra;
      if (MODE == 3)      ra = row_map[base + (rr < Me ? rr : (Me - 1))];
      else if (MODE == 4) ra = base + (rr < Me ? rr : (Me - 1));
      else                ra = rr;
      *(v4u*)(smA + r * 40 + c4 * 8) =
          *(const v4u*)(A + (long long)ra * Kd + k0 + c4 * 8);
    }
#if USE_TDM
    if (tid == 0) tdm_w_tile(W + (long long)n0 * Kd + k0, (void*)smW, Kd, N);
#else
#pragma unroll
    for (int u = tid; u < 512; u += 256) {
      int r = u >> 2, c4 = u & 3;
      *(v4u*)(smW + r * 40 + c4 * 8) =
          *(const v4u*)(W + (long long)(n0 + r) * Kd + k0 + c4 * 8);
    }
#endif
    __syncthreads();
    v16bf a0 = ld_fragA(smA + (wm + 0) * 40, 40, lane);
    v16bf a1 = ld_fragA(smA + (wm + 16) * 40, 40, lane);
#pragma unroll
    for (int j = 0; j < 4; ++j) {
      v16bf bw = ld_fragB(smW + (wn + j * 16) * 40, 40, lane);
      acc[0][j] = wmma_bf16(a0, bw, acc[0][j]);
      acc[1][j] = wmma_bf16(a1, bw, acc[1][j]);
    }
  }

  // epilogue: C layout row = 8*(lane>>4)+t, col = lane&15
  const int crow = (lane >> 4) * 8;
  const int ccol = lane & 15;
#pragma unroll
  for (int i = 0; i < 2; ++i)
#pragma unroll
    for (int j = 0; j < 4; ++j)
#pragma unroll
      for (int t = 0; t < 8; ++t) {
        int gm = m0 + wm + i * 16 + crow + t;
        int gn = n0 + wn + j * 16 + ccol;
        float v = acc[i][j][t];
        if (MODE == 0) {
          v += bias[gn];
          outF[(long long)gm * N + gn] = v;
        } else if (MODE == 1) {
          v += bias[gn];
          int bb = gm >> 10, tt = gm & 1023;
          if (gn < kD) {
            int hh = gn >> 6, dd = gn & 63;
            outB[(((long long)bb * kH + hh) * kT + tt) * kHD + dd] = (__bf16)v;
          } else {
            int c = gn - kD, hh = c >> 6, dd = c & 63;
            outB2[(((long long)bb * kH + hh) * kT + tt) * kHD + dd] = (__bf16)v;
          }
        } else if (MODE == 2) {
          v += bias[gn];
          int bb = gm >> 10, tt = gm & 1023;
          int hh = gn >> 6, dd = gn & 63;
          outB[(((long long)bb * kH + hh) * kHD + dd) * kT + tt] = (__bf16)v;
        } else if (MODE == 3) {
          if (gm < Me) {
            v += bias[gn];
            v = v / (1.f + __expf(-v));  // SiLU
            outB[(long long)(base + gm) * N + gn] = (__bf16)v;
          }
        } else {  // MODE == 4
          if (gm < Me) {
            v += bias[gn];
            outF[(long long)(base + gm) * N + gn] = v;
          }
        }
      }
}

// ---------------- banded flash attention ----------------
// grid: (T/128, B*H); 8 waves, each owns 16 query rows. q,k: [b,h,t,hd]; vt: [b,h,hd,t]
__global__ __launch_bounds__(256) void k_attn(const __bf16* __restrict__ q,
                                              const __bf16* __restrict__ kmat,
                                              const __bf16* __restrict__ vt,
                                              __bf16* __restrict__ ctx) {
  const int bh = blockIdx.y;
  const int qs = blockIdx.x * 128;
  const int tid = threadIdx.x, lane = tid & 31, wv = tid >> 5;
  const int q0 = qs + wv * 16;
  __shared__ __bf16 smK[64 * 72];
  __shared__ __bf16 smV[64 * 72];
  __shared__ __bf16 smP[8 * 16 * 72];
  __bf16* pw = smP + wv * (16 * 72);

  // persistent Q fragments (2 K-chunks of 32 over HD=64)
  v16bf aq0, aq1;
  {
    const __bf16* p = q + ((long long)bh * kT + q0 + (lane & 15)) * kHD + ((lane >> 4) << 3);
    Frag32 f;
    f.a = *(const v4u*)(p);      f.b = *(const v4u*)(p + 16);
    aq0 = __builtin_bit_cast(v16bf, f);
    f.a = *(const v4u*)(p + 32); f.b = *(const v4u*)(p + 48);
    aq1 = __builtin_bit_cast(v16bf, f);
  }

  float mrun[8], lrun[8];
  v8f o[4];
#pragma unroll
  for (int i = 0; i < 8; ++i) { mrun[i] = -1e30f; lrun[i] = 0.f; }
#pragma unroll
  for (int j = 0; j < 4; ++j) o[j] = vzero();

  int kb0 = qs - kBand + 1;
  if (kb0 < 0) kb0 = 0;
  kb0 &= ~63;
  const int crow = (lane >> 4) * 8;
  const int ccol = lane & 15;

  for (int kb = kb0; kb < qs + 128; kb += 64) {
    __syncthreads();
#pragma unroll
    for (int u = tid; u < 512; u += 256) {
      int r = u >> 3, c4 = u & 7;
      *(v4u*)(smK + r * 72 + c4 * 8) =
          *(const v4u*)(kmat + ((long long)bh * kT + kb + r) * kHD + c4 * 8);
      *(v4u*)(smV + r * 72 + c4 * 8) =
          *(const v4u*)(vt + ((long long)bh * kHD + r) * kT + kb + c4 * 8);
    }
    __syncthreads();

    // S = Q K^T  (16 x 64 per wave)
    v8f s[4];
#pragma unroll
    for (int j = 0; j < 4; ++j) {
      v8f z = vzero();
      z = wmma_bf16(aq0, ld_fragB(smK + j * 16 * 72, 72, lane), z);
      z = wmma_bf16(aq1, ld_fragB(smK + j * 16 * 72 + 32, 72, lane), z);
      s[j] = z;
    }
    // mask + scale + block row-max
    float bm[8];
#pragma unroll
    for (int i = 0; i < 8; ++i) bm[i] = -1e30f;
#pragma unroll
    for (int j = 0; j < 4; ++j)
#pragma unroll
      for (int i = 0; i < 8; ++i) {
        int qr = q0 + crow + i;
        int jj = kb + j * 16 + ccol;
        float v = s[j][i] * 0.125f;  // 1/sqrt(64)
        v = (jj <= qr && jj > qr - kBand) ? v : -1e30f;
        s[j][i] = v;
        bm[i] = fmaxf(bm[i], v);
      }
#pragma unroll
    for (int m = 1; m < 16; m <<= 1)
#pragma unroll
      for (int i = 0; i < 8; ++i) bm[i] = fmaxf(bm[i], __shfl_xor(bm[i], m));
    // online softmax state update
    float scl[8], ls[8];
#pragma unroll
    for (int i = 0; i < 8; ++i) {
      float mn = fmaxf(mrun[i], bm[i]);
      scl[i] = __expf(mrun[i] - mn);
      mrun[i] = mn;
      ls[i] = 0.f;
    }
#pragma unroll
    for (int j = 0; j < 4; ++j)
#pragma unroll
      for (int i = 0; i < 8; ++i) {
        float p = __expf(s[j][i] - mrun[i]);
        s[j][i] = p;
        ls[i] += p;
      }
#pragma unroll
    for (int m = 1; m < 16; m <<= 1)
#pragma unroll
      for (int i = 0; i < 8; ++i) ls[i] += __shfl_xor(ls[i], m);
#pragma unroll
    for (int i = 0; i < 8; ++i) lrun[i] = lrun[i] * scl[i] + ls[i];
#pragma unroll
    for (int j = 0; j < 4; ++j)
#pragma unroll
      for (int i = 0; i < 8; ++i) o[j][i] *= scl[i];
    // P -> LDS (C-layout to A-layout round trip)
#pragma unroll
    for (int j = 0; j < 4; ++j)
#pragma unroll
      for (int i = 0; i < 8; ++i)
        pw[(crow + i) * 72 + j * 16 + ccol] = (__bf16)s[j][i];
    __syncthreads();
    // O += P * V   (V^T fragments from vt tile)
    v16bf ap0 = ld_fragA(pw, 72, lane);
    v16bf ap1 = ld_fragA(pw + 32, 72, lane);
#pragma unroll
    for (int j = 0; j < 4; ++j) {
      o[j] = wmma_bf16(ap0, ld_fragB(smV + j * 16 * 72, 72, lane), o[j]);
      o[j] = wmma_bf16(ap1, ld_fragB(smV + j * 16 * 72 + 32, 72, lane), o[j]);
    }
  }

  const int b = bh >> 4, h = bh & 15;
#pragma unroll
  for (int i = 0; i < 8; ++i) {
    float inv = 1.f / lrun[i];
    int t = q0 + crow + i;
#pragma unroll
    for (int j = 0; j < 4; ++j)
      ctx[(((long long)b * kT + t) * kH + h) * kHD + j * 16 + ccol] =
          (__bf16)(o[j][i] * inv);
  }
}

// ---------------- router: top-2 of 8, softmax gates ----------------
__global__ __launch_bounds__(256) void k_router(const __bf16* __restrict__ x2,
                                                const float* __restrict__ rw,
                                                const float* __restrict__ rb,
                                                float* __restrict__ gates,
                                                int* __restrict__ experts,
                                                int* __restrict__ cnt) {
  const int tok = blockIdx.x;
  const int lane = threadIdx.x & 31, wv = threadIdx.x >> 5;
  const __bf16* xr = x2 + (long long)tok * kD;
  const float* wr = rw + wv * kD;
  float s = 0.f;
  for (int i = lane; i < kD; i += 32) s += (float)xr[i] * wr[i];
#pragma unroll
  for (int m = 16; m; m >>= 1) s += __shfl_xor(s, m);
  __shared__ float lg[kE];
  if (lane == 0) lg[wv] = s + rb[wv];
  __syncthreads();
  if (threadIdx.x == 0) {
    int i1 = 0;
    for (int e = 1; e < kE; ++e)
      if (lg[e] > lg[i1]) i1 = e;
    int i2 = -1;
    for (int e = 0; e < kE; ++e)
      if (e != i1 && (i2 < 0 || lg[e] > lg[i2])) i2 = e;
    float e2 = __expf(lg[i2] - lg[i1]);
    float inv = 1.f / (1.f + e2);
    gates[tok * 2 + 0] = inv;
    gates[tok * 2 + 1] = e2 * inv;
    experts[tok * 2 + 0] = i1;
    experts[tok * 2 + 1] = i2;
    atomicAdd(&cnt[i1], 1);
    atomicAdd(&cnt[i2], 1);
  }
}

__global__ void k_scan(const int* __restrict__ cnt, int* __restrict__ basearr) {
  if (threadIdx.x == 0) {
    int acc = 0;
    for (int e = 0; e < kE; ++e) { basearr[e] = acc; acc += cnt[e]; }
    basearr[kE] = acc;
  }
}

__global__ __launch_bounds__(256) void k_assign(const int* __restrict__ experts,
                                                const int* __restrict__ basearr,
                                                int* __restrict__ cursor,
                                                int* __restrict__ row_map,
                                                int* __restrict__ slots) {
  int tok = blockIdx.x * 256 + threadIdx.x;
  if (tok >= kTok) return;
#pragma unroll
  for (int kk = 0; kk < 2; ++kk) {
    int e = experts[tok * 2 + kk];
    int pos = atomicAdd(&cursor[e], 1);
    int slot = basearr[e] + pos;
    row_map[slot] = tok;
    slots[tok * 2 + kk] = slot;
  }
}

// ---------------- final combine: out = attn_out + g0*y0 + g1*y1 ----------------
__global__ __launch_bounds__(256) void k_final(const float* __restrict__ attn,
                                               const float* __restrict__ ybuf,
                                               const float* __restrict__ gates,
                                               const int* __restrict__ slots,
                                               float* __restrict__ out) {
  const int tok = blockIdx.x;
  const float* y0 = ybuf + (long long)slots[tok * 2 + 0] * kD;
  const float* y1 = ybuf + (long long)slots[tok * 2 + 1] * kD;
  const float g0 = gates[tok * 2 + 0], g1 = gates[tok * 2 + 1];
  for (int i = threadIdx.x * 4; i < kD; i += 1024) {
    float4 a = *(const float4*)(attn + (long long)tok * kD + i);
    float4 p0 = *(const float4*)(y0 + i);
    float4 p1 = *(const float4*)(y1 + i);
    float4 r;
    r.x = a.x + g0 * p0.x + g1 * p1.x;
    r.y = a.y + g0 * p0.y + g1 * p1.y;
    r.z = a.z + g0 * p0.z + g1 * p1.z;
    r.w = a.w + g0 * p0.w + g1 * p1.w;
    *(float4*)(out + (long long)tok * kD + i) = r;
  }
}

// ---------------- orchestration ----------------
extern "C" void kernel_launch(void* const* d_in, const int* in_sizes, int n_in,
                              void* d_out, int out_size, void* d_ws, size_t ws_size,
                              hipStream_t stream) {
  (void)in_sizes; (void)n_in; (void)out_size; (void)ws_size;
  const float* x    = (const float*)d_in[0];
  const float* rs   = (const float*)d_in[1];
  const float* wqkv = (const float*)d_in[2];
  const float* bqkv = (const float*)d_in[3];
  const float* wo   = (const float*)d_in[4];
  const float* bo   = (const float*)d_in[5];
  const float* rw   = (const float*)d_in[6];
  const float* rb   = (const float*)d_in[7];
  const float* w1   = (const float*)d_in[8];
  const float* b1   = (const float*)d_in[9];
  const float* w2   = (const float*)d_in[10];
  const float* b2   = (const float*)d_in[11];
  float* out = (float*)d_out;

  char* p = (char*)d_ws;
  auto take = [&](size_t bytes) -> void* {
    char* r = p;
    p += (bytes + 255) & ~(size_t)255;
    return (void*)r;
  };
  __bf16* wqkvB = (__bf16*)take((size_t)3 * kD * kD * 2);
  __bf16* woB   = (__bf16*)take((size_t)kD * kD * 2);
  __bf16* w1B   = (__bf16*)take((size_t)kE * kDFF * kD * 2);
  __bf16* w2B   = (__bf16*)take((size_t)kE * kD * kDFF * 2);
  __bf16* xnB   = (__bf16*)take((size_t)kTok * kD * 2);
  __bf16* qkB   = (__bf16*)take((size_t)kTok * kD * 2);
  __bf16* qB    = (__bf16*)take((size_t)kTok * kD * 2);
  __bf16* kB    = (__bf16*)take((size_t)kTok * kD * 2);
  __bf16* vtB   = (__bf16*)take((size_t)kTok * kD * 2);
  __bf16* ctxB  = (__bf16*)take((size_t)kTok * kD * 2);
  __bf16* x2B   = (__bf16*)take((size_t)kTok * kD * 2);
  __bf16* hB    = (__bf16*)take((size_t)kRows * kDFF * 2);
  float* attn   = (float*)take((size_t)kTok * kD * 4);
  float* ybuf   = (float*)take((size_t)kRows * kD * 4);
  float* gates  = (float*)take((size_t)kTok * 2 * 4);
  int* experts  = (int*)take((size_t)kTok * 2 * 4);
  int* slots    = (int*)take((size_t)kTok * 2 * 4);
  int* rowmap   = (int*)take((size_t)kRows * 4);
  int* cnt      = (int*)take(16 * 4);  // cnt[8] + cursor[8]
  int* basearr  = (int*)take(16 * 4);
  int* cursor   = cnt + 8;

  (void)hipMemsetAsync(cnt, 0, 16 * sizeof(int), stream);

  k_cvt<<<(3 * kD * kD) / 1024, 256, 0, stream>>>(wqkv, wqkvB, 3 * kD * kD);
  k_cvt<<<(kD * kD) / 1024, 256, 0, stream>>>(wo, woB, kD * kD);
  k_cvt<<<(kE * kDFF * kD) / 1024, 256, 0, stream>>>(w1, w1B, kE * kDFF * kD);
  k_cvt<<<(kE * kD * kDFF) / 1024, 256, 0, stream>>>(w2, w2B, kE * kD * kDFF);

  k_rms<true><<<kTok, 256, 0, stream>>>(x, rs, xnB, qkB);

  // Q,K from rope'd input; V (transposed) from plain normed input
  k_gemm<1><<<dim3(kTok / 128, (2 * kD) / 128, 1), 256, 0, stream>>>(
      qkB, wqkvB, bqkv, nullptr, qB, kB, kTok, 2 * kD, kD,
      nullptr, nullptr, nullptr, 0);
  k_gemm<2><<<dim3(kTok / 128, kD / 128, 1), 256, 0, stream>>>(
      xnB, wqkvB + (size_t)2 * kD * kD, bqkv + 2 * kD, nullptr, vtB, nullptr,
      kTok, kD, kD, nullptr, nullptr, nullptr, 0);

  k_attn<<<dim3(kT / 128, kBatch * kH), 256, 0, stream>>>(qB, kB, vtB, ctxB);

  k_gemm<0><<<dim3(kTok / 128, kD / 128, 1), 256, 0, stream>>>(
      ctxB, woB, bo, attn, nullptr, nullptr, kTok, kD, kD,
      nullptr, nullptr, nullptr, 0);

  k_rms<false><<<kTok, 256, 0, stream>>>(attn, rs, x2B, nullptr);

  k_router<<<kTok, 256, 0, stream>>>(x2B, rw, rb, gates, experts, cnt);
  k_scan<<<1, 32, 0, stream>>>(cnt, basearr);
  k_assign<<<kTok / 256, 256, 0, stream>>>(experts, basearr, cursor, rowmap, slots);

  k_gemm<3><<<dim3(kTok / 128, kDFF / 128, kE), 256, 0, stream>>>(
      x2B, w1B, b1, nullptr, hB, nullptr, kTok, kDFF, kD,
      rowmap, basearr, cnt, (long long)kDFF * kD);
  k_gemm<4><<<dim3(kTok / 128, kD / 128, kE), 256, 0, stream>>>(
      hB, w2B, b2, ybuf, nullptr, nullptr, kTok, kD, kDFF,
      nullptr, basearr, cnt, (long long)kD * kDFF);

  k_final<<<kTok, 256, 0, stream>>>(attn, ybuf, gates, slots, out);
}
